// TransformerEncoderBidirectional_62002147885088
// MI455X (gfx1250) — matrix operand order repaired and checked by reading, not compiled, not run on hardware
//
#include <hip/hip_runtime.h>
#include <math.h>

// Shapes (fixed by the reference)
#define B_   4
#define SL_  1024
#define S_   1026        // 1 (bos) + 1 (soft prompt) + 1024
#define D_   512
#define H_   8
#define DH_  64
#define L_   6
#define FF_  2048
#define M_   (B_ * S_)   // 4104 flattened rows

typedef __attribute__((ext_vector_type(16))) __bf16 v16bf;
typedef __attribute__((ext_vector_type(8)))  float  v8f;

static __device__ __forceinline__ float gelu_exact(float x) {
  return 0.5f * x * (1.0f + erff(x * 0.7071067811865475f));
}

// ---------------------------------------------------------------------------
// Embed: x[b,0]=bos, x[b,1]=soft_prompt, x[b,2+t]=concat(loss_proj, gap_proj)
// Also builds additive mask (0 or -1e9) of shape (B, S).
// ---------------------------------------------------------------------------
__global__ void embed_kernel(const float* __restrict__ losses,
                             const float* __restrict__ gaps,
                             const float* __restrict__ attn_mask,
                             const float* __restrict__ soft_prompts,
                             const float* __restrict__ lossW,
                             const float* __restrict__ lossb,
                             const float* __restrict__ gapW,
                             const float* __restrict__ gapb,
                             const float* __restrict__ bos,
                             float* __restrict__ x,
                             float* __restrict__ maskadd) {
  int idx = blockIdx.x * blockDim.x + threadIdx.x;
  if (idx >= B_ * S_ * D_) return;
  int d = idx % D_;
  int s = (idx / D_) % S_;
  int b = idx / (D_ * S_);
  float val;
  if (s == 0) {
    val = bos[d];
  } else if (s == 1) {
    val = soft_prompts[b * D_ + d];
  } else {
    int t = s - 2;
    if (d < D_ / 2) val = losses[b * SL_ + t] * lossW[d] + lossb[d];
    else            val = gaps[b * SL_ + t] * gapW[d - D_ / 2] + gapb[d - D_ / 2];
  }
  x[idx] = val;
  if (d == 0) {
    float m = (s < 2) ? 1.f : attn_mask[b * SL_ + (s - 2)];
    maskadd[b * S_ + s] = (m > 0.f) ? 0.f : -1e9f;
  }
}

// ---------------------------------------------------------------------------
// Pack weights: Wp[n][k] = bf16(W[k][n]) -- B-operand fragment-friendly layout
// ---------------------------------------------------------------------------
__global__ void pack_w_kernel(const float* __restrict__ W,
                              __bf16* __restrict__ Wp, int K, int N) {
  int idx = blockIdx.x * blockDim.x + threadIdx.x;
  if (idx >= K * N) return;
  int k = idx % K;        // output-coalesced along k
  int n = idx / K;
  Wp[(size_t)n * K + k] = (__bf16)W[(size_t)k * N + n];
}

// ---------------------------------------------------------------------------
// LayerNorm over D=512; one 128-thread block per row, 4 floats per thread.
// ---------------------------------------------------------------------------
__global__ __launch_bounds__(128) void ln_kernel(const float* __restrict__ in,
                                                 const float* __restrict__ g,
                                                 const float* __restrict__ be,
                                                 float* __restrict__ out,
                                                 size_t in_row_stride,
                                                 size_t out_row_stride) {
  __shared__ float red[4];
  int row = blockIdx.x;
  int tid = threadIdx.x;
  const float* ip = in + (size_t)row * in_row_stride;
  float* op = out + (size_t)row * out_row_stride;
  int c = tid * 4;
  float4 x = *(const float4*)(ip + c);
  float s = x.x + x.y + x.z + x.w;
#pragma unroll
  for (int m = 1; m < 32; m <<= 1) s += __shfl_xor(s, m, 32);
  if ((tid & 31) == 0) red[tid >> 5] = s;
  __syncthreads();
  s = red[0] + red[1] + red[2] + red[3];
  float mu = s * (1.0f / (float)D_);
  float d0 = x.x - mu, d1 = x.y - mu, d2 = x.z - mu, d3 = x.w - mu;
  float v = d0 * d0 + d1 * d1 + d2 * d2 + d3 * d3;
#pragma unroll
  for (int m = 1; m < 32; m <<= 1) v += __shfl_xor(v, m, 32);
  __syncthreads();
  if ((tid & 31) == 0) red[tid >> 5] = v;
  __syncthreads();
  v = (red[0] + red[1] + red[2] + red[3]) * (1.0f / (float)D_);
  float rs = rsqrtf(v + 1e-5f);
  float4 gg = *(const float4*)(g + c);
  float4 bb = *(const float4*)(be + c);
  float4 o4;
  o4.x = d0 * rs * gg.x + bb.x;
  o4.y = d1 * rs * gg.y + bb.y;
  o4.z = d2 * rs * gg.z + bb.z;
  o4.w = d3 * rs * gg.w + bb.w;
  *(float4*)(op + c) = o4;
}

// ---------------------------------------------------------------------------
// RoPE applied in-place to q and k.
// ---------------------------------------------------------------------------
__global__ void rope_kernel(float* __restrict__ q, float* __restrict__ k) {
  int idx = blockIdx.x * blockDim.x + threadIdx.x;
  const int total = B_ * S_ * H_ * (DH_ / 2);
  if (idx >= total) return;
  int i = idx & 31;
  int hh = (idx >> 5) & (H_ - 1);
  int s = (idx >> 8) % S_;
  int b = idx / (256 * S_);
  float inv = expf(-(float)i * (9.210340371976184f / 32.0f)); // 10000^(-i/32)
  float ang = (float)s * inv;
  float sn, cs;
  sincosf(ang, &sn, &cs);
  size_t base = ((size_t)(b * S_ + s)) * D_ + hh * DH_;
  float q1 = q[base + i], q2 = q[base + i + 32];
  q[base + i]      = q1 * cs - q2 * sn;
  q[base + i + 32] = q2 * cs + q1 * sn;
  float k1 = k[base + i], k2 = k[base + i + 32];
  k[base + i]      = k1 * cs - k2 * sn;
  k[base + i + 32] = k2 * cs + k1 * sn;
}

// ---------------------------------------------------------------------------
// WMMA GEMM: out[M,N] = act(A[M,K] @ W[K,N] + bias) (+ resid)
// A: fp32 row-major (cvt on fragment build).
// Wp: bf16, pre-packed [n][k] -- staged to LDS via GLOBAL_LOAD_ASYNC_TO_LDS.
// Block = 128 threads (4 waves, 2x2), 64x64 tile, k-step 32.
// ---------------------------------------------------------------------------
__global__ __launch_bounds__(128) void gemm_kernel(const float* __restrict__ A,
                                                   const __bf16* __restrict__ Wp,
                                                   const float* __restrict__ bias,
                                                   const float* __restrict__ resid,
                                                   float* __restrict__ out,
                                                   int M, int K, int N, int act) {
  __shared__ float  As[64][36];    // [m][k], stride 36 floats (16B-aligned rows)
  __shared__ __bf16 Bsb[64][48];   // [n][k], stride 48 bf16 = 96B (16B-aligned)
  int tid = threadIdx.x;
  int lane = tid & 31, wave = tid >> 5;
  int hi = lane >> 4, lm = lane & 15;
  int wm = (wave >> 1) * 32, wn = (wave & 1) * 32;
  int bm = blockIdx.y * 64, bn = blockIdx.x * 64;

  v8f acc[2][2] = {};
  for (int k0 = 0; k0 < K; k0 += 32) {
    // --- async-stage bf16 B tile: 64 n-rows x 32 k = 256 x 16B units ------
    {
      const __bf16* wbase = Wp + (size_t)bn * K + k0;   // uniform SGPR base
#pragma unroll
      for (int i = 0; i < 2; ++i) {
        int u = tid + 128 * i;          // 0..255
        int n = u >> 2;
        int koff = (u & 3) * 8;         // bf16 elements
        unsigned voff = (unsigned)((n * K + koff) * 2); // byte offset
        unsigned ldsa = (unsigned)(unsigned long long)(const void*)&Bsb[n][koff];
        asm volatile("global_load_async_to_lds_b128 %0, %1, %2"
                     :: "v"(ldsa), "v"(voff), "s"(wbase)
                     : "memory");
      }
    }
    // --- stage A 64x32 fp32 (overlaps with async B) -----------------------
#pragma unroll
    for (int i = 0; i < 4; ++i) {
      int f = tid + 128 * i;
      int r = f >> 3, c4 = (f & 7) * 4;
      int gr = bm + r;
      float4 val = make_float4(0.f, 0.f, 0.f, 0.f);
      if (gr < M) {
        val = *(const float4*)(A + (size_t)gr * K + k0 + c4);
        if (i == 0 && k0 + 32 < K)
          __builtin_prefetch(A + (size_t)gr * K + k0 + 32 + c4, 0, 0);
      }
      *(float4*)(&As[r][c4]) = val;
    }
    asm volatile("s_wait_asynccnt 0x0" ::: "memory");
    __syncthreads();

    // --- build fragments --------------------------------------------------
    v16bf af[2], bfr[2];
#pragma unroll
    for (int mt = 0; mt < 2; ++mt) {
      int m = wm + mt * 16 + lm;
#pragma unroll
      for (int e = 0; e < 16; e += 2) {
        int vv = e >> 1;
        int kk = hi * 16 + vv * 2;
        float2 f2 = *(const float2*)(&As[m][kk]);
        af[mt][e] = (__bf16)f2.x;
        af[mt][e + 1] = (__bf16)f2.y;
      }
    }
#pragma unroll
    for (int nt = 0; nt < 2; ++nt)
      bfr[nt] = *(const v16bf*)(&Bsb[wn + nt * 16 + lm][hi * 16]);

#pragma unroll
    for (int mt = 0; mt < 2; ++mt)
#pragma unroll
      for (int nt = 0; nt < 2; ++nt)
        acc[mt][nt] = __builtin_amdgcn_wmma_f32_16x16x32_bf16(
            false, af[mt], false, bfr[nt], (short)0, acc[mt][nt], false, false);
    __syncthreads();
  }

  // Epilogue: bias, activation, residual, guarded store
#pragma unroll
  for (int mt = 0; mt < 2; ++mt)
#pragma unroll
    for (int nt = 0; nt < 2; ++nt) {
      int col = bn + wn + nt * 16 + lm;
      float bv = bias ? bias[col] : 0.f;
#pragma unroll
      for (int r = 0; r < 8; ++r) {
        int row = bm + wm + mt * 16 + r + hi * 8;
        if (row < M) {
          float v = acc[mt][nt][r] + bv;
          if (act == 1) v = gelu_exact(v);
          if (resid) v += resid[(size_t)row * N + col];
          out[(size_t)row * N + col] = v;
        }
      }
    }
}

// ---------------------------------------------------------------------------
// Flash-style attention: grid (ceil(S/64), H, B); 4 waves, each owns 16 q rows.
// (K/V staged with guarded fp32 loads: the ragged last tile needs zero-fill
//  and V needs a transpose, which async copies cannot do.)
// ---------------------------------------------------------------------------
__global__ __launch_bounds__(128) void attn_kernel(const float* __restrict__ q,
                                                   const float* __restrict__ k,
                                                   const float* __restrict__ v,
                                                   const float* __restrict__ maskadd,
                                                   float* __restrict__ o) {
  __shared__ float Ks[32][68];       // [key_row][d]
  __shared__ float Vs[64][36];       // transposed: [d][key_row]
  __shared__ float Ps[4][16][34];    // per-wave P tile [m][key_col]

  int tid = threadIdx.x, lane = tid & 31, wave = tid >> 5;
  int hi = lane >> 4, lm = lane & 15;
  int b = blockIdx.z, h = blockIdx.y;
  int q0 = blockIdx.x * 64 + wave * 16;

  const float* qg = q + ((size_t)b * S_) * D_ + h * DH_;
  const float* kg = k + ((size_t)b * S_) * D_ + h * DH_;
  const float* vg = v + ((size_t)b * S_) * D_ + h * DH_;

  // Preload this wave's q fragments (2 k-steps over DH=64)
  v16bf aq[2];
  int qr = q0 + lm;
  if (qr >= S_) qr = S_ - 1;
#pragma unroll
  for (int t = 0; t < 2; ++t)
#pragma unroll
    for (int e = 0; e < 16; e += 2) {
      int vv = e >> 1;
      int c = t * 32 + hi * 16 + vv * 2;
      float2 f2 = *(const float2*)(qg + (size_t)qr * D_ + c);
      aq[t][e] = (__bf16)f2.x;
      aq[t][e + 1] = (__bf16)f2.y;
    }

  float mrun[8], lrun[8];
  v8f oacc[4] = {};
#pragma unroll
  for (int r = 0; r < 8; ++r) { mrun[r] = -__builtin_inff(); lrun[r] = 0.f; }
  const float scale = 0.125f;  // DH^-0.5

  int nblocks = (S_ + 31) / 32;
  for (int kb = 0; kb < nblocks; ++kb) {
    int kr0 = kb * 32;
#pragma unroll
    for (int i = 0; i < 4; ++i) {
      int f = tid + 128 * i;
      int r = f >> 4, c4 = (f & 15) * 4;
      int gr = kr0 + r;
      float4 kk = make_float4(0.f, 0.f, 0.f, 0.f);
      float4 vv = make_float4(0.f, 0.f, 0.f, 0.f);
      if (gr < S_) {
        kk = *(const float4*)(kg + (size_t)gr * D_ + c4);
        vv = *(const float4*)(vg + (size_t)gr * D_ + c4);
      }
      *(float4*)(&Ks[r][c4]) = kk;
      Vs[c4 + 0][r] = vv.x;
      Vs[c4 + 1][r] = vv.y;
      Vs[c4 + 2][r] = vv.z;
      Vs[c4 + 3][r] = vv.w;
    }
    __syncthreads();

    // scores tile 16x32 = 2 accumulators, 2 k-steps each
    v8f sc[2] = {};
#pragma unroll
    for (int t = 0; t < 2; ++t)
#pragma unroll
      for (int u = 0; u < 2; ++u) {
        v16bf bk;
        const float* kp = &Ks[u * 16 + lm][t * 32 + hi * 16];
#pragma unroll
        for (int e = 0; e < 16; e += 4) {
          float4 f4 = *(const float4*)(kp + e);
          bk[e] = (__bf16)f4.x;
          bk[e + 1] = (__bf16)f4.y;
          bk[e + 2] = (__bf16)f4.z;
          bk[e + 3] = (__bf16)f4.w;
        }
        sc[u] = __builtin_amdgcn_wmma_f32_16x16x32_bf16(
            false, aq[t], false, bk, (short)0, sc[u], false, false);
      }

    int c0 = kr0 + lm, c1 = kr0 + 16 + lm;
    float msk0 = (c0 < S_) ? maskadd[b * S_ + c0] : -__builtin_inff();
    float msk1 = (c1 < S_) ? maskadd[b * S_ + c1] : -__builtin_inff();

    // Online softmax per row (rows live in 16-lane halves)
#pragma unroll
    for (int r = 0; r < 8; ++r) {
      float s0 = sc[0][r] * scale + msk0;
      float s1 = sc[1][r] * scale + msk1;
      float vm = fmaxf(s0, s1);
#pragma unroll
      for (int x = 1; x < 16; x <<= 1) vm = fmaxf(vm, __shfl_xor(vm, x, 32));
      float mnew = fmaxf(mrun[r], vm);
      float corr = __expf(mrun[r] - mnew);
      float e0 = __expf(s0 - mnew);
      float e1 = __expf(s1 - mnew);
      float rs = e0 + e1;
#pragma unroll
      for (int x = 1; x < 16; x <<= 1) rs += __shfl_xor(rs, x, 32);
      lrun[r] = lrun[r] * corr + rs;
      mrun[r] = mnew;
#pragma unroll
      for (int u = 0; u < 4; ++u) oacc[u][r] *= corr;
      Ps[wave][r + hi * 8][lm] = e0;
      Ps[wave][r + hi * 8][16 + lm] = e1;
    }

    // PV: A = P (16x32) via per-wave LDS transpose; B = V (32x64), 4 n-tiles
    v16bf ap;
#pragma unroll
    for (int e = 0; e < 16; e += 2) {
      int vv = e >> 1;
      int c = hi * 16 + vv * 2;
      float2 f2 = *(const float2*)(&Ps[wave][lm][c]);
      ap[e] = (__bf16)f2.x;
      ap[e + 1] = (__bf16)f2.y;
    }
#pragma unroll
    for (int u = 0; u < 4; ++u) {
      v16bf bv;
      const float* vp = &Vs[u * 16 + lm][hi * 16];
#pragma unroll
      for (int e = 0; e < 16; e += 4) {
        float4 f4 = *(const float4*)(vp + e);
        bv[e] = (__bf16)f4.x;
        bv[e + 1] = (__bf16)f4.y;
        bv[e + 2] = (__bf16)f4.z;
        bv[e + 3] = (__bf16)f4.w;
      }
      oacc[u] = __builtin_amdgcn_wmma_f32_16x16x32_bf16(
          false, ap, false, bv, (short)0, oacc[u], false, false);
    }
    __syncthreads();
  }

  // Normalize and store
#pragma unroll
  for (int u = 0; u < 4; ++u)
#pragma unroll
    for (int r = 0; r < 8; ++r) {
      int row = q0 + r + hi * 8;
      if (row < S_) {
        float val = oacc[u][r] / lrun[r];
        o[((size_t)b * S_ + row) * D_ + h * DH_ + u * 16 + lm] = val;
      }
    }
}

// ---------------------------------------------------------------------------
// Launcher
// ---------------------------------------------------------------------------
extern "C" void kernel_launch(void* const* d_in, const int* in_sizes, int n_in,
                              void* d_out, int out_size, void* d_ws, size_t ws_size,
                              hipStream_t stream) {
  (void)in_sizes; (void)n_in; (void)out_size; (void)ws_size;
  const float* losses        = (const float*)d_in[0];
  const float* gaps          = (const float*)d_in[1];
  const float* attention_mask= (const float*)d_in[2];
  const float* soft_prompts  = (const float*)d_in[3];
  const float* loss_W        = (const float*)d_in[4];
  const float* loss_b        = (const float*)d_in[5];
  const float* gap_W         = (const float*)d_in[6];
  const float* gap_b         = (const float*)d_in[7];
  const float* bos           = (const float*)d_in[8];
  const float* Wq            = (const float*)d_in[9];
  const float* bq            = (const float*)d_in[10];
  const float* Wk            = (const float*)d_in[11];
  const float* bk            = (const float*)d_in[12];
  const float* Wv            = (const float*)d_in[13];
  const float* bv            = (const float*)d_in[14];
  const float* Wo            = (const float*)d_in[15];
  const float* bo            = (const float*)d_in[16];
  const float* W1            = (const float*)d_in[17];
  const float* b1            = (const float*)d_in[18];
  const float* W2            = (const float*)d_in[19];
  const float* b2            = (const float*)d_in[20];
  const float* ln1_g         = (const float*)d_in[21];
  const float* ln1_b         = (const float*)d_in[22];
  const float* ln2_g         = (const float*)d_in[23];
  const float* ln2_b         = (const float*)d_in[24];
  const float* fn_g          = (const float*)d_in[25];
  const float* fn_b          = (const float*)d_in[26];

  float* ws = (float*)d_ws;
  const size_t N1 = (size_t)B_ * S_ * D_;
  float* xbuf = ws;
  float* hbuf = xbuf + N1;
  float* qbuf = hbuf + N1;
  float* kbuf = qbuf + N1;
  float* vbuf = kbuf + N1;
  float* obuf = vbuf + N1;
  float* ffbuf = qbuf;               // FF_ = 4*D_, aliases q,k,v,o (free then)
  float* maskadd = obuf + N1;
  __bf16* packW = (__bf16*)(maskadd + (size_t)B_ * S_);  // D_*FF_ bf16 max

  // Build x and mask
  {
    int total = B_ * S_ * D_;
    embed_kernel<<<(total + 255) / 256, 256, 0, stream>>>(
        losses, gaps, attention_mask, soft_prompts, loss_W, loss_b,
        gap_W, gap_b, bos, xbuf, maskadd);
  }

  const int Mtiles = (M_ + 63) / 64;
  const dim3 gemmQKV((D_ / 64), Mtiles);
  const dim3 gemmFF1((FF_ / 64), Mtiles);
  const dim3 attnGrid((S_ + 63) / 64, H_, B_);
  const int ropeTotal = B_ * S_ * H_ * (DH_ / 2);
  const int packDD = (D_ * D_ + 255) / 256;
  const int packDF = (D_ * FF_ + 255) / 256;

  for (int l = 0; l < L_; ++l) {
    const float* Wq_l = Wq + (size_t)l * D_ * D_;
    const float* Wk_l = Wk + (size_t)l * D_ * D_;
    const float* Wv_l = Wv + (size_t)l * D_ * D_;
    const float* Wo_l = Wo + (size_t)l * D_ * D_;
    const float* W1_l = W1 + (size_t)l * D_ * FF_;
    const float* W2_l = W2 + (size_t)l * FF_ * D_;

    ln_kernel<<<M_, 128, 0, stream>>>(xbuf, ln1_g + l * D_, ln1_b + l * D_,
                                      hbuf, D_, D_);
    pack_w_kernel<<<packDD, 256, 0, stream>>>(Wq_l, packW, D_, D_);
    gemm_kernel<<<gemmQKV, 128, 0, stream>>>(hbuf, packW, bq + l * D_, nullptr,
                                             qbuf, M_, D_, D_, 0);
    pack_w_kernel<<<packDD, 256, 0, stream>>>(Wk_l, packW, D_, D_);
    gemm_kernel<<<gemmQKV, 128, 0, stream>>>(hbuf, packW, bk + l * D_, nullptr,
                                             kbuf, M_, D_, D_, 0);
    pack_w_kernel<<<packDD, 256, 0, stream>>>(Wv_l, packW, D_, D_);
    gemm_kernel<<<gemmQKV, 128, 0, stream>>>(hbuf, packW, bv + l * D_, nullptr,
                                             vbuf, M_, D_, D_, 0);
    rope_kernel<<<(ropeTotal + 255) / 256, 256, 0, stream>>>(qbuf, kbuf);
    attn_kernel<<<attnGrid, 128, 0, stream>>>(qbuf, kbuf, vbuf, maskadd, obuf);
    pack_w_kernel<<<packDD, 256, 0, stream>>>(Wo_l, packW, D_, D_);
    gemm_kernel<<<gemmQKV, 128, 0, stream>>>(obuf, packW, bo + l * D_, xbuf,
                                             xbuf, M_, D_, D_, 0);
    ln_kernel<<<M_, 128, 0, stream>>>(xbuf, ln2_g + l * D_, ln2_b + l * D_,
                                      hbuf, D_, D_);
    pack_w_kernel<<<packDF, 256, 0, stream>>>(W1_l, packW, D_, FF_);
    gemm_kernel<<<gemmFF1, 128, 0, stream>>>(hbuf, packW, b1 + l * FF_, nullptr,
                                             ffbuf, M_, D_, FF_, 1);
    pack_w_kernel<<<packDF, 256, 0, stream>>>(W2_l, packW, FF_, D_);
    gemm_kernel<<<gemmQKV, 128, 0, stream>>>(ffbuf, packW, b2 + l * D_, xbuf,
                                             xbuf, M_, FF_, D_, 0);
  }

  // Final LN on token 0 of each batch -> d_out (B, D)
  ln_kernel<<<B_, 128, 0, stream>>>(xbuf, fn_g, fn_b, (float*)d_out,
                                    (size_t)S_ * D_, (size_t)D_);
}
